// DINet_36378372997680
// MI455X (gfx1250) — matrix operand order repaired and checked by reading, not compiled, not run on hardware
//
#include <hip/hip_runtime.h>

// ---------------------------------------------------------------------------
// DINet cosine-correlation matcher for MI455X (gfx1250, wave32, WMMA).
// B=2, C=64, H=W=64, L=4096, patch dim C*9=576.
//   K1: unfold 3x3 + L2-normalize -> f16 Kn/Qn [B][L][576]
//   K2: fused f16 WMMA GEMM (Kn x Qn^T) + max/argmax over K axis
//       - 32 q-columns per block (2 N-tiles/wave) => each A fragment feeds
//         2 WMMAs, halving L2->WGP traffic per FLOP
//       - Qn strip staged to LDS via GLOBAL_LOAD_ASYNC_TO_LDS_B128 (ASYNCcnt)
//   K3: gather matched V patches + fold (overlap-add)/9
// ---------------------------------------------------------------------------

typedef __attribute__((ext_vector_type(16))) _Float16 v16h;
typedef __attribute__((ext_vector_type(8)))  float    v8f;
typedef __attribute__((ext_vector_type(4)))  unsigned int uint4v;

union HalfPack { uint4v u[2]; v16h h; };

#define BATCH 2
#define LL    4096
#define C9    576
#define QSTRIDE 584   // padded LDS row stride (halves): 1168B -> conflict-free b128

// ---------------------------------------------------------------------------
// Kernel 1: unfold(3x3, pad 1) + L2 normalize, one wave (32 lanes) per patch.
// ---------------------------------------------------------------------------
__global__ __launch_bounds__(256) void unfold_norm_kernel(
    const float* __restrict__ Ksrc, const float* __restrict__ Qsrc,
    _Float16* __restrict__ Kn, _Float16* __restrict__ Qn)
{
    int tid  = threadIdx.x;
    int lane = tid & 31;
    int wave = blockIdx.x * 8 + (tid >> 5);   // 0..16383
    int sel  = wave >> 13;                    // 0 = K, 1 = Q
    int b    = (wave >> 12) & 1;
    int l    = wave & 4095;
    int ly = l >> 6, lx = l & 63;

    const float* src = sel ? Qsrc : Ksrc;
    _Float16*    dst = sel ? Qn   : Kn;

    float vals[18];
    float ss = 0.f;
#pragma unroll
    for (int t = 0; t < 18; ++t) {
        int idx = lane + (t << 5);            // 0..575 = c*9 + i*3 + j
        int c = idx / 9;
        int r = idx - c * 9;
        int i = r / 3;
        int j = r - i * 3;
        int y = ly + i - 1, x = lx + j - 1;
        float v = 0.f;
        if ((unsigned)y < 64u && (unsigned)x < 64u)
            v = src[(((b << 6) + c) << 12) + (y << 6) + x];
        vals[t] = v;
        ss += v * v;
    }
#pragma unroll
    for (int off = 16; off > 0; off >>= 1)
        ss += __shfl_xor(ss, off, 32);
    float scale = 1.f / fmaxf(sqrtf(ss), 1e-12f);   // matches F.normalize eps

    _Float16* drow = dst + (size_t)((b << 12) + l) * C9;
#pragma unroll
    for (int t = 0; t < 18; ++t)
        drow[lane + (t << 5)] = (_Float16)(vals[t] * scale);
}

// ---------------------------------------------------------------------------
// Kernel 2: fused correlation GEMM + max/argmax over k.
// grid = (128 q-blocks, B); block = 256 threads = 8 wave32.
// Block owns 32 q columns (2 N-tiles). Each wave sweeps 32 of 256 M-blocks;
// per M-block: 18 K-steps, each 1 A-load feeding 2 WMMAs.
// ---------------------------------------------------------------------------
__global__ __launch_bounds__(256) void corr_max_kernel(
    const _Float16* __restrict__ Kn, const _Float16* __restrict__ Qn,
    float* __restrict__ S, int* __restrict__ Arg)
{
    __shared__ _Float16 qtile[32 * QSTRIDE];
    __shared__ float redv[2][256];
    __shared__ int   redk[2][256];

    int tid   = threadIdx.x;
    int qbase = blockIdx.x << 5;             // 32 columns per block
    int b     = blockIdx.y;

    // ---- Stage 32 Qn rows (576 halves each) into LDS (padded stride) via
    //      async global->LDS DMA (ASYNCcnt path). 32*72 = 2304 b128 chunks.
    const _Float16* qsrc = Qn + ((size_t)(b << 12) + qbase) * C9;
    unsigned qtile_base = (unsigned)(uintptr_t)(&qtile[0]);  // LDS byte offset
#pragma unroll 1
    for (int t = tid; t < 32 * 72; t += 256) {
        int row = t / 72;
        int col = (t - row * 72) << 3;                        // halves, mult of 8
        const _Float16* g = qsrc + (size_t)row * C9 + col;
        unsigned ldsoff = qtile_base + (unsigned)(row * QSTRIDE + col) * 2u;
        asm volatile("global_load_async_to_lds_b128 %0, %1, off"
                     :: "v"(ldsoff), "v"(g) : "memory");
    }
    asm volatile("s_wait_asynccnt 0" ::: "memory");
    __syncthreads();

    int wave = tid >> 5, lane = tid & 31;
    int half = lane >> 4, mrem = lane & 15;

    float best0 = -3.4e38f, best1 = -3.4e38f;
    int   bk0 = 0, bk1 = 0;

    const _Float16* Abase = Kn + (size_t)(b << 12) * C9;
    // B fragments: lane -> column mrem; halves 0/1 take K 0-7/16-23 vs 8-15/24-31.
    const _Float16* brow0 = &qtile[mrem * QSTRIDE + (half << 3)];
    const _Float16* brow1 = brow0 + 16 * QSTRIDE;

    for (int mb = wave; mb < 256; mb += 8) {
        v8f acc0 = {0.f, 0.f, 0.f, 0.f, 0.f, 0.f, 0.f, 0.f};
        v8f acc1 = {0.f, 0.f, 0.f, 0.f, 0.f, 0.f, 0.f, 0.f};
        // A fragment: lane -> row mrem; halves 0/1 take K 0-7/16-23 vs 8-15/24-31.
        const _Float16* arow =
            Abase + (size_t)((mb << 4) + mrem) * C9 + (half << 3);
#pragma unroll
        for (int ks = 0; ks < 18; ++ks) {
            int kb = ks << 5;
            HalfPack A, B0, B1;
            A.u[0]  = *(const uint4v*)(arow + kb);
            A.u[1]  = *(const uint4v*)(arow + kb + 16);
            B0.u[0] = *(const uint4v*)(brow0 + kb);
            B0.u[1] = *(const uint4v*)(brow0 + kb + 16);
            B1.u[0] = *(const uint4v*)(brow1 + kb);
            B1.u[1] = *(const uint4v*)(brow1 + kb + 16);
            acc0 = __builtin_amdgcn_wmma_f32_16x16x32_f16(
                false, A.h, false, B0.h, (short)0, acc0, false, false);
            acc1 = __builtin_amdgcn_wmma_f32_16x16x32_f16(
                false, A.h, false, B1.h, (short)0, acc1, false, false);
        }
        // C layout: VGPR r -> row (mb*16 + half*8 + r), column mrem.
        int krow0 = (mb << 4) + (half << 3);
#pragma unroll
        for (int r = 0; r < 8; ++r) {
            float v0 = acc0[r];
            if (v0 > best0) { best0 = v0; bk0 = krow0 + r; }  // ascending k
            float v1 = acc1[r];
            if (v1 > best1) { best1 = v1; bk1 = krow0 + r; }
        }
    }

    redv[0][tid] = best0; redk[0][tid] = bk0;
    redv[1][tid] = best1; redk[1][tid] = bk1;
    __syncthreads();

    if (tid < 32) {   // one thread per q column (2 tiles x 16 cols)
        int t0 = tid >> 4;        // q-tile
        int n  = tid & 15;        // column within tile
        float bv = -3.4e38f;
        int   bk = 0x7fffffff;
        for (int w = 0; w < 8; ++w)
            for (int h = 0; h < 2; ++h) {
                int e = w * 32 + (h << 4) + n;
                float v = redv[t0][e];
                int   k = redk[t0][e];
                if (v > bv || (v == bv && k < bk)) { bv = v; bk = k; }
            }
        int q = qbase + (t0 << 4) + n;
        S[(b << 12) + q]   = bv;
        Arg[(b << 12) + q] = bk;
    }
}

// ---------------------------------------------------------------------------
// Kernel 3: T[b,c,y,x] = (1/9) * sum_{i,j} Vpad[b,c, ky(q)+i, kx(q)+j]
// over valid q=(y+1-i, x+1-j); matched location from Arg.
// ---------------------------------------------------------------------------
__global__ __launch_bounds__(256) void gather_fold_kernel(
    const float* __restrict__ V, const int* __restrict__ Arg,
    float* __restrict__ T)
{
    int g = blockIdx.x * 256 + threadIdx.x;          // < B*C*H*W = 524288
    int x = g & 63, y = (g >> 6) & 63, c = (g >> 12) & 63, b = g >> 18;
    const float* Vb   = V + ((size_t)((b << 6) + c) << 12);
    const int*   argb = Arg + (b << 12);

    float s = 0.f;
#pragma unroll
    for (int i = 0; i < 3; ++i)
#pragma unroll
        for (int j = 0; j < 3; ++j) {
            int qy = y + 1 - i, qx = x + 1 - j;
            if ((unsigned)qy < 64u && (unsigned)qx < 64u) {
                int k  = argb[(qy << 6) + qx];
                int vy = (k >> 6) + i - 1;
                int vx = (k & 63) + j - 1;
                if ((unsigned)vy < 64u && (unsigned)vx < 64u)
                    s += Vb[(vy << 6) + vx];
            }
        }
    T[g] = s * (1.f / 9.f);
}

// ---------------------------------------------------------------------------
extern "C" void kernel_launch(void* const* d_in, const int* in_sizes, int n_in,
                              void* d_out, int out_size, void* d_ws, size_t ws_size,
                              hipStream_t stream)
{
    (void)in_sizes; (void)n_in; (void)out_size; (void)ws_size;

    const float* V = (const float*)d_in[0];
    const float* K = (const float*)d_in[1];
    const float* Q = (const float*)d_in[2];

    float* S = (float*)d_out;                 // [B,1,H,W] = 8192 floats
    float* T = S + (size_t)BATCH * LL;        // [B,C,H,W] = 524288 floats

    // Workspace: Kn (9.44 MB) | Qn (9.44 MB) | Arg (32 KB)
    _Float16* Kn = (_Float16*)d_ws;
    _Float16* Qn = Kn + (size_t)BATCH * LL * C9;
    int*      Arg = (int*)(Qn + (size_t)BATCH * LL * C9);

    // K1: 2 tensors * B * L patches = 16384 waves = 2048 blocks of 8 waves.
    unfold_norm_kernel<<<2048, 256, 0, stream>>>(K, Q, Kn, Qn);

    // K2: 128 q-blocks x B batches, 256 threads (8 wave32) each.
    corr_max_kernel<<<dim3(128, BATCH), 256, 0, stream>>>(Kn, Qn, S, Arg);

    // K3: 524288 output elements.
    gather_fold_kernel<<<2048, 256, 0, stream>>>(V, Arg, T);
}
